// RIPU_Net_73272142069891
// MI455X (gfx1250) — matrix operand order repaired and checked by reading, not compiled, not run on hardware
//
#include <hip/hip_runtime.h>
#include <hip/hip_bf16.h>

typedef __attribute__((ext_vector_type(16))) _Float16 v16h;
typedef __attribute__((ext_vector_type(4)))  _Float16 v4h;
typedef __attribute__((ext_vector_type(8)))  float    v8f;
typedef __attribute__((ext_vector_type(4)))  float    v4f;

#define B_   4
#define C_   19
#define H_   512
#define W_   1024
#define HW_  (H_*W_)
#define R_   4            // radius for SIZE=9
#define HALO 24           // 16 + 2*R_
#define HROW 32           // padded halo rows (24..31 = sentinel) -> K reads unconditional
#define TSTR 32           // staging row stride (cols 24..31 zeroed)
#define EPS_ 1e-6f
#define LOGC 2.9444389791664403f  // log(19)

// ---------------------------------------------------------------------------
// Pass 1: per-pixel argmax class (u8) + unnormalized pixel entropy (f16).
// 4 pixels per thread via b128 non-temporal loads (single-use 160MB stream).
// ---------------------------------------------------------------------------
__global__ __launch_bounds__(256) void ripu_prep(const v4f* __restrict__ pred4,
                                                 unsigned int* __restrict__ cls4,
                                                 v4h* __restrict__ ent4) {
  int q = blockIdx.x * 256 + threadIdx.x;         // quad index
  if (q >= B_ * (HW_ / 4)) return;
  int b   = q / (HW_ / 4);
  int hw4 = q - b * (HW_ / 4);
  const v4f* p = pred4 + (size_t)b * C_ * (HW_ / 4) + hw4;

  float maxv[4] = {-1.f, -1.f, -1.f, -1.f};
  int   maxi[4] = {0, 0, 0, 0};
  float ent[4]  = {0.f, 0.f, 0.f, 0.f};
#pragma unroll
  for (int c = 0; c < C_; ++c) {
    v4f v = __builtin_nontemporal_load(p + (size_t)c * (HW_ / 4));
#pragma unroll
    for (int j = 0; j < 4; ++j) {
      float vj = v[j];
      ent[j] -= vj * __logf(vj + EPS_);
      if (vj > maxv[j]) { maxv[j] = vj; maxi[j] = c; }
    }
  }
  cls4[q] = (unsigned)maxi[0] | ((unsigned)maxi[1] << 8) |
            ((unsigned)maxi[2] << 16) | ((unsigned)maxi[3] << 24);
  v4h e;
  e[0] = (_Float16)ent[0]; e[1] = (_Float16)ent[1];
  e[2] = (_Float16)ent[2]; e[3] = (_Float16)ent[3];
  ent4[q] = e;
}

// ---------------------------------------------------------------------------
// Pass 2: 9x9 box filters as banded WMMA matmuls.
// ONE wave32 per workgroup => barriers are S_NOP (single-wave WG), LDS private.
// ---------------------------------------------------------------------------
struct WaveLds {
  unsigned char cls[HROW * HALO];   // halo class map, rows 24..31 = 255
  _Float16      ent[HROW * HALO];   // halo pixel entropy, rows 24..31 = 0
  _Float16      tst[16 * TSTR];     // vertical-sum staging, cols 24..31 = 0
};

__device__ __forceinline__ v8f wmma_f16(v16h a, v16h b, v8f c) {
  return __builtin_amdgcn_wmma_f32_16x16x32_f16(false, a, false, b,
                                                (short)0, c, false, false);
}

__global__ __launch_bounds__(32) void ripu_score(const unsigned char* __restrict__ clsmap,
                                                 const _Float16* __restrict__ entmap,
                                                 const int* __restrict__ use_entropy,
                                                 float* __restrict__ out) {
  __shared__ WaveLds L;

  const int lane = threadIdx.x;     // 0..31
  const int half = lane >> 4;       // lane group 0/1
  const int lr   = lane & 15;

  // 8192 tiles: 64 across, 32 down, 4 batches; one tile per (single-wave) block
  const int tile = blockIdx.x;
  const int tx = tile & 63;
  const int ty = (tile >> 6) & 31;
  const int b  = tile >> 11;
  const int x0 = tx * 16, y0 = ty * 16;
  const size_t ibase = (size_t)b * HW_;

  // ---- halo fill (24x24 real, branch-free via clamp+select): 18 iters ----
#pragma unroll 3
  for (int idx = lane; idx < HALO * HALO; idx += 32) {
    int row = idx / HALO, col = idx - row * HALO;
    int gy = y0 - R_ + row, gx = x0 - R_ + col;
    bool ok = (gy >= 0) && (gy < H_) && (gx >= 0) && (gx < W_);
    int cy = min(max(gy, 0), H_ - 1);
    int cx = min(max(gx, 0), W_ - 1);
    size_t g = ibase + (size_t)cy * W_ + cx;
    unsigned char cv = clsmap[g];
    _Float16      ev = entmap[g];
    L.cls[idx] = ok ? cv : (unsigned char)255;
    L.ent[idx] = ok ? ev : (_Float16)0.0f;
  }
  // ---- pad rows 24..31 (sentinel class / zero entropy): 6 iters ----
#pragma unroll
  for (int idx = lane; idx < 8 * HALO; idx += 32) {
    int o = HALO * HALO + idx;
    L.cls[o] = (unsigned char)255;
    L.ent[o] = (_Float16)0.0f;
  }
  // ---- zero staging tail cols 24..31: 4 iters ----
#pragma unroll
  for (int idx = lane; idx < 16 * 8; idx += 32) {
    int row = idx >> 3, col = 24 + (idx & 7);
    L.tst[row * TSTR + col] = (_Float16)0.0f;
  }
  __syncthreads();   // single-wave WG: S_NOP, orders DS only

  // ---- banded summation operands (constants, built once) ----
  // A 16-bit layout: lane row = lr; half i -> K = half*16+i.  A[y][k] = (y<=k<=y+8)
  // B 16-bit layout: lane col = lr; half i -> K = half*16+i. Bh[k][x] = (x<=k<=x+8)
  v16h Aband, Bhband;
#pragma unroll
  for (int i = 0; i < 16; ++i) {
    int ka = half * 16 + i;
    Aband[i]  = (_Float16)((ka < HALO && ka >= lr && ka <= lr + 8) ? 1.0f : 0.0f);
    int kb = half * 16 + i;
    Bhband[i] = (_Float16)((kb < HALO && kb >= lr && kb <= lr + 8) ? 1.0f : 0.0f);
  }

  // ---- hoist B-operand source data into registers (read LDS once) ----
  v16h Kc0, Kc1, Be0, Be1;
#pragma unroll
  for (int i = 0; i < 16; ++i) {
    int K = half * 16 + i;
    Kc0[i] = (_Float16)(float)L.cls[K * HALO + lr];
    Kc1[i] = (_Float16)(float)L.cls[K * HALO + lr + 8];
    Be0[i] = L.ent[K * HALO + lr];
    Be1[i] = L.ent[K * HALO + lr + 8];
  }

  // Separable 9x9 box filter: vertical (2 col groups) then horizontal.
  auto boxpair = [&](v16h Bv0, v16h Bv1) -> v8f {
    v8f zero = {};
    v8f T0 = wmma_f16(Aband, Bv0, zero);
    v8f T1 = wmma_f16(Aband, Bv1, zero);
    // C-layout -> staging rows: VGPR i = rows i / i+8
#pragma unroll
    for (int i = 0; i < 8; ++i) {
      L.tst[(i + half * 8) * TSTR + lr]     = (_Float16)T0[i];
      L.tst[(i + half * 8) * TSTR + lr + 8] = (_Float16)T1[i];
    }
    __syncthreads();
    v16h Ta;
#pragma unroll
    for (int i = 0; i < 16; ++i) {
      int k = half * 16 + i;
      Ta[i] = L.tst[lr * TSTR + k];
    }
    v8f res = wmma_f16(Ta, Bhband, zero);
    __syncthreads();   // staging reused by next call
    return res;
  };

  const _Float16 h1 = (_Float16)1.0f, h0 = (_Float16)0.0f;
  const _Float16 sent = (_Float16)255.0f;

  // total = box of valid-pixel mask (exact SAME zero-padding semantics)
  v16h M0, M1;
#pragma unroll
  for (int i = 0; i < 16; ++i) {
    M0[i] = (Kc0[i] == sent) ? h0 : h1;
    M1[i] = (Kc1[i] == sent) ? h0 : h1;
  }
  v8f Tot  = boxpair(M0, M1);
  // region-summed pixel entropy
  v8f Ebox = boxpair(Be0, Be1);

  float invT[8], entAcc[8];
#pragma unroll
  for (int i = 0; i < 8; ++i) {
    invT[i]   = 1.0f / Tot[i];
    entAcc[i] = 0.0f;
  }

  // per-class local counts -> region impurity entropy (compares only, no DS)
#pragma unroll 1
  for (int c = 0; c < C_; ++c) {
    _Float16 hc = (_Float16)(float)c;
    v16h B0, B1;
#pragma unroll
    for (int i = 0; i < 16; ++i) {
      B0[i] = (Kc0[i] == hc) ? h1 : h0;
      B1[i] = (Kc1[i] == hc) ? h1 : h0;
    }
    v8f Cnt = boxpair(B0, B1);
#pragma unroll
    for (int i = 0; i < 8; ++i) {
      float p = Cnt[i] * invT[i];
      entAcc[i] -= p * __logf(p + EPS_);
    }
  }

  const float inv_logC = 1.0f / LOGC;
  const int ue = use_entropy[0];
#pragma unroll
  for (int i = 0; i < 8; ++i) {
    float s = entAcc[i] * inv_logC;
    if (ue) s *= Ebox[i] * invT[i] * inv_logC;
    int row = i + half * 8, col = lr;
    __builtin_nontemporal_store(s, &out[ibase + (size_t)(y0 + row) * W_ + (x0 + col)]);
  }
}

// ---------------------------------------------------------------------------
extern "C" void kernel_launch(void* const* d_in, const int* in_sizes, int n_in,
                              void* d_out, int out_size, void* d_ws, size_t ws_size,
                              hipStream_t stream) {
  const float* pred = (const float*)d_in[0];
  const int*   ue   = (const int*)d_in[1];
  float*       out  = (float*)d_out;

  // workspace layout: [0,2MB) class map u8, [4MB,8MB) entropy f16
  unsigned char* clsbuf = (unsigned char*)d_ws;
  _Float16*      entbuf = (_Float16*)((char*)d_ws + (size_t)(4u << 20));

  const int nquad = B_ * (HW_ / 4);
  ripu_prep<<<(nquad + 255) / 256, 256, 0, stream>>>(
      (const v4f*)pred, (unsigned int*)clsbuf, (v4h*)entbuf);

  const int ntiles = B_ * (H_ / 16) * (W_ / 16);  // 8192
  ripu_score<<<ntiles, 32, 0, stream>>>(clsbuf, entbuf, ue, out);
}
